// Transformer_146028888655
// MI455X (gfx1250) — matrix operand order repaired and checked by reading, not compiled, not run on hardware
//
#include <hip/hip_runtime.h>
#include <math.h>

typedef __bf16 bf16_t;
typedef __attribute__((ext_vector_type(8)))  __bf16 v8bf;
typedef __attribute__((ext_vector_type(16))) __bf16 v16bf;
typedef __attribute__((ext_vector_type(8)))  float  v8f;

#define DEV static __device__ __forceinline__

constexpr int Bc = 8, Nc = 4096, Dc = 256, Hc = 8, DHc = 32, Mc = 256, HIDc = 1024;
constexpr int ROWS = Bc * Nc;                       // 32768
constexpr float NRM = 0.4204482076268573f;          // 32^(-1/4)

// ---------- helpers ----------
DEV float bf2f(bf16_t h) {
  unsigned short s = __builtin_bit_cast(unsigned short, h);
  unsigned u = ((unsigned)s) << 16;
  return __builtin_bit_cast(float, u);
}
DEV bf16_t f2bf(float f) {
  unsigned u = __builtin_bit_cast(unsigned, f);
  unsigned r = u + 0x7FFFu + ((u >> 16) & 1u);      // RNE
  return __builtin_bit_cast(bf16_t, (unsigned short)(r >> 16));
}
DEV v8f z8() {
  v8f z;
#pragma unroll
  for (int i = 0; i < 8; i++) z[i] = 0.f;
  return z;
}
DEV v8f wmma_bf(v16bf a, v16bf b, v8f c) {
  return __builtin_amdgcn_wmma_f32_16x16x32_bf16(false, a, false, b, (short)0, c, false, false);
}
// A fragment: LDS row-major [16 rows][K], lane L holds row L&15, K = (L>>4)*8 + 0..7 and +16
DEV v16bf ldfragA(const bf16_t* base, int strideE, int lane) {
  int r = lane & 15, hf = lane >> 4;
  v8bf lo = *(const v8bf*)(base + r * strideE + hf * 8);
  v8bf hi = *(const v8bf*)(base + r * strideE + 16 + hf * 8);
  v16bf o;
#pragma unroll
  for (int i = 0; i < 8; i++) { o[i] = lo[i]; o[8 + i] = hi[i]; }
  return o;
}
// B fragment from transposed LDS Bt[col][k]: lane L -> col L&15, K = (L>>4)*8 + 0..7 and +16
DEV v16bf ldfragB(const bf16_t* base, int strideE, int lane) {
  int c = lane & 15, hf = lane >> 4;
  v8bf lo = *(const v8bf*)(base + c * strideE + hf * 8);
  v8bf hi = *(const v8bf*)(base + c * strideE + 16 + hf * 8);
  v16bf o;
#pragma unroll
  for (int i = 0; i < 8; i++) { o[i] = lo[i]; o[8 + i] = hi[i]; }
  return o;
}
DEV float gelu_f(float x) { return 0.5f * x * (1.0f + erff(x * 0.7071067811865475f)); }

// CDNA5 async global->LDS copy (ASYNCcnt tracked). vdst = per-lane LDS byte
// offset (low 32 bits of generic pointer == logical LDS address, §10.2).
DEV unsigned lds_off_u32(const void* p) { return (unsigned)(size_t)p; }
DEV void async_copy_b128(unsigned lds_byte, const bf16_t* g) {
  asm volatile("global_load_async_to_lds_b128 %0, %1, off"
               :: "v"(lds_byte), "v"(g) : "memory");
}
DEV void async_wait0() { asm volatile("s_wait_asynccnt 0" ::: "memory"); }

// ---------- elementwise kernels ----------
__global__ void k_cvt(const float* __restrict__ s, bf16_t* __restrict__ d, int n) {
  int i = blockIdx.x * 256 + threadIdx.x;
  if (i < n) d[i] = f2bf(s[i]);
}
__global__ void k_addpos(const float* __restrict__ x, const float* __restrict__ p,
                         float* __restrict__ h) {
  int i = blockIdx.x * 256 + threadIdx.x;
  h[i] = x[i] + p[i % (Nc * Dc)];
}
__global__ void k_copy(const float* __restrict__ s, float* __restrict__ d) {
  int i = blockIdx.x * 256 + threadIdx.x;
  d[i] = s[i];
}
__global__ void k_zero_umax(unsigned* u) { u[threadIdx.x] = 0u; }

// ---------- layernorm (1 row / 256-thread block) ----------
__global__ __launch_bounds__(256)
void k_ln(const float* __restrict__ h, const float* __restrict__ g,
          const float* __restrict__ bta, bf16_t* __restrict__ o) {
  __shared__ float red[8];
  const int row = blockIdx.x, t = threadIdx.x;
  float v = h[(size_t)row * Dc + t];
  float s = v;
#pragma unroll
  for (int k = 16; k > 0; k >>= 1) s += __shfl_xor(s, k, 32);
  if ((t & 31) == 0) red[t >> 5] = s;
  __syncthreads();
  float mu = 0.f;
#pragma unroll
  for (int i = 0; i < 8; i++) mu += red[i];
  mu *= (1.0f / 256.0f);
  __syncthreads();
  float d = v - mu;
  float s2 = d * d;
#pragma unroll
  for (int k = 16; k > 0; k >>= 1) s2 += __shfl_xor(s2, k, 32);
  if ((t & 31) == 0) red[t >> 5] = s2;
  __syncthreads();
  float var = 0.f;
#pragma unroll
  for (int i = 0; i < 8; i++) var += red[i];
  var *= (1.0f / 256.0f);
  float rs = rsqrtf(var + 1e-6f);
  o[(size_t)row * Dc + t] = f2bf(d * rs * g[t] + bta[t]);
}

// ---------- generic bf16 WMMA GEMM, 128x64 workgroup tile ----------
// ACT: 0=none 1=gelu ; MODE: 0=store bf16 to outB, 1=residual add into outF (f32)
template <int ACT, int MODE>
__global__ __launch_bounds__(256)
void k_gemm(const bf16_t* __restrict__ A, const bf16_t* __restrict__ W,
            const float* __restrict__ bias, float* __restrict__ outF,
            bf16_t* __restrict__ outB, int K, int Ncols) {
  constexpr int AS = 40;  // stride (elems); 80B = 16B multiple
  __shared__ __attribute__((aligned(16))) bf16_t lA[128 * AS];
  __shared__ __attribute__((aligned(16))) bf16_t lBt[64 * AS];
  const int t = threadIdx.x, lane = t & 31, w = t >> 5;
  const int m0 = blockIdx.x * 128, n0 = blockIdx.y * 64;
  const int wc = (w & 3) * 16, wr = (w >> 2) * 64;   // wave: 1 col subtile, 4 row subtiles
  const int ar = t >> 1, ak = (t & 1) * 16;          // A stage: 16 elems (32B) per thread
  const int bk = t >> 3, bn = (t & 7) * 8;           // W stage: 8 elems per thread (transpose)
  const unsigned ldsA0 = lds_off_u32(lA + ar * AS + ak);
  v8f acc[4];
#pragma unroll
  for (int i = 0; i < 4; i++) acc[i] = z8();
  for (int k0 = 0; k0 < K; k0 += 32) {
    __syncthreads();
    const bf16_t* ga = A + (size_t)(m0 + ar) * K + k0 + ak;
    async_copy_b128(ldsA0, ga);            // async DMA into LDS (ASYNCcnt)
    async_copy_b128(ldsA0 + 16, ga + 8);
    v8bf wv = *(const v8bf*)(W + (size_t)(k0 + bk) * Ncols + n0 + bn);
#pragma unroll
    for (int i = 0; i < 8; i++) lBt[(bn + i) * AS + bk] = wv[i];
    if (k0 + 32 < K)
      __builtin_prefetch(W + (size_t)(k0 + 32 + bk) * Ncols + n0 + bn, 0, 0);
    async_wait0();
    __syncthreads();
    v16bf bfr = ldfragB(lBt + wc * AS, AS, lane);
    v16bf a0 = ldfragA(lA + (wr + 0) * AS, AS, lane);
    v16bf a1 = ldfragA(lA + (wr + 16) * AS, AS, lane);
    v16bf a2 = ldfragA(lA + (wr + 32) * AS, AS, lane);
    v16bf a3 = ldfragA(lA + (wr + 48) * AS, AS, lane);
    acc[0] = wmma_bf(a0, bfr, acc[0]);
    acc[1] = wmma_bf(a1, bfr, acc[1]);
    acc[2] = wmma_bf(a2, bfr, acc[2]);
    acc[3] = wmma_bf(a3, bfr, acc[3]);
  }
  const int col = n0 + wc + (lane & 15);
  const int hf = lane >> 4;
  const float bv = bias ? bias[col] : 0.0f;
#pragma unroll
  for (int s = 0; s < 4; s++) {
#pragma unroll
    for (int v = 0; v < 8; v++) {
      int r = m0 + wr + s * 16 + hf * 8 + v;
      float xv = acc[s][v] + bv;
      if (ACT == 1) xv = gelu_f(xv);
      size_t idx = (size_t)r * Ncols + col;
      if (MODE == 0) outB[idx] = f2bf(xv);
      else           outF[idx] += xv;
    }
  }
}

// ---------- per-head global max of u = (k*nrm) @ proj^T ----------
__global__ __launch_bounds__(256)
void k_umax(const bf16_t* __restrict__ qkv, const bf16_t* __restrict__ pw,
            unsigned* __restrict__ umax) {
  constexpr int PS = 32;
  __shared__ __attribute__((aligned(16))) bf16_t lP[256 * PS];
  __shared__ __attribute__((aligned(16))) bf16_t lA[64 * PS];
  __shared__ float red[8];
  const int t = threadIdx.x, lane = t & 31, w = t >> 5;
  const int bh = blockIdx.y, b = bh >> 3, h = bh & 7;
  const int n0 = blockIdx.x * 64;
#pragma unroll
  for (int i = 0; i < 4; i++)
    *(v8bf*)(lP + t * PS + i * 8) = *(const v8bf*)(pw + t * PS + i * 8);
  {
    int r = t >> 2, kg = (t & 3) * 8;
    v8bf kv = *(const v8bf*)(qkv + (size_t)(b * Nc + n0 + r) * 768 + 256 + h * 32 + kg);
    v8bf o;
#pragma unroll
    for (int i = 0; i < 8; i++) o[i] = f2bf(bf2f(kv[i]) * NRM);
    *(v8bf*)(lA + r * PS + kg) = o;
  }
  __syncthreads();
  float mym = -3.0e38f;
  const int rt = (w >> 1) * 16, cb = (w & 1) * 128;
  v16bf af = ldfragA(lA + rt * PS, PS, lane);
#pragma unroll
  for (int ct = 0; ct < 8; ct++) {
    v16bf bf = ldfragB(lP + (cb + ct * 16) * PS, PS, lane);
    v8f u = wmma_bf(af, bf, z8());
#pragma unroll
    for (int v = 0; v < 8; v++) mym = fmaxf(mym, u[v]);
  }
#pragma unroll
  for (int o = 16; o > 0; o >>= 1) mym = fmaxf(mym, __shfl_xor(mym, o, 32));
  if (lane == 0) red[w] = mym;
  __syncthreads();
  if (t == 0) {
    float m = red[0];
#pragma unroll
    for (int i = 1; i < 8; i++) m = fmaxf(m, red[i]);
    unsigned ub = __builtin_bit_cast(unsigned, m);
    unsigned key = (ub & 0x80000000u) ? ~ub : (ub | 0x80000000u);
    atomicMax(umax + bh, key);  // order-independent -> deterministic
  }
}

// ---------- kp = phi(k); kv = kp^T @ v ; ksum = sum_n kp  (per-chunk slices, no atomics) ----------
__global__ __launch_bounds__(256)
void k_kvsum(const bf16_t* __restrict__ qkv, const bf16_t* __restrict__ pw,
             const unsigned* __restrict__ umaxb,
             float* __restrict__ kvb, float* __restrict__ ksumb) {
  constexpr int PS = 32, VS = 72, KS = 72;
  __shared__ __attribute__((aligned(16))) bf16_t lP[256 * PS];
  __shared__ __attribute__((aligned(16))) bf16_t lA[64 * PS];
  __shared__ __attribute__((aligned(16))) bf16_t lVt[32 * VS];
  __shared__ __attribute__((aligned(16))) bf16_t lKpT[256 * KS];
  __shared__ float lDiag[64];
  const int t = threadIdx.x, lane = t & 31, w = t >> 5;
  const int bh = blockIdx.y, b = bh >> 3, h = bh & 7;
  const int nbase = blockIdx.x * 1024;
  unsigned key = umaxb[bh];
  unsigned mbits = (key & 0x80000000u) ? (key & 0x7FFFFFFFu) : ~key;
  const float mx = __builtin_bit_cast(float, mbits);
#pragma unroll
  for (int i = 0; i < 4; i++)
    *(v8bf*)(lP + t * PS + i * 8) = *(const v8bf*)(pw + t * PS + i * 8);
  v8f acc[4];
#pragma unroll
  for (int i = 0; i < 4; i++) acc[i] = z8();
  float ksacc = 0.f;
  const int rt = (w >> 1) * 16, cb = (w & 1) * 128;
  const int r4 = t >> 2, g8 = (t & 3) * 8;
  for (int it = 0; it < 16; ++it) {
    const int n0 = nbase + it * 64;
    __syncthreads();
    {
      v8bf kv = *(const v8bf*)(qkv + (size_t)(b * Nc + n0 + r4) * 768 + 256 + h * 32 + g8);
      v8bf o;
#pragma unroll
      for (int i = 0; i < 8; i++) o[i] = f2bf(bf2f(kv[i]) * NRM);
      *(v8bf*)(lA + r4 * PS + g8) = o;
      v8bf vv = *(const v8bf*)(qkv + (size_t)(b * Nc + n0 + r4) * 768 + 512 + h * 32 + g8);
#pragma unroll
      for (int i = 0; i < 8; i++) lVt[(g8 + i) * VS + r4] = vv[i];
    }
    __syncthreads();
    if (t < 64) {
      float s = 0.f;
      for (int d = 0; d < 32; d++) { float xx = bf2f(lA[t * PS + d]); s += xx * xx; }
      lDiag[t] = 0.5f * s;
    }
    __syncthreads();
    {
      v16bf af = ldfragA(lA + rt * PS, PS, lane);
      const int hf = lane >> 4;
#pragma unroll
      for (int ct = 0; ct < 8; ct++) {
        const int mbase = cb + ct * 16;
        v16bf bf = ldfragB(lP + mbase * PS, PS, lane);
        v8f u = wmma_bf(af, bf, z8());
        const int mcol = mbase + (lane & 15);
#pragma unroll
        for (int v = 0; v < 8; v++) {
          int rl = rt + hf * 8 + v;
          float kp = (expf(u[v] - lDiag[rl] - mx) + 1e-4f) * 0.0625f;
          lKpT[mcol * KS + rl] = f2bf(kp);
        }
      }
    }
    __syncthreads();
    {
      float s = 0.f;
      for (int j = 0; j < 64; j++) s += bf2f(lKpT[t * KS + j]);
      ksacc += s;
    }
    {
      const int mrow = w * 32;
#pragma unroll
      for (int ks = 0; ks < 2; ks++) {
        v16bf a0 = ldfragA(lKpT + mrow * KS + ks * 32, KS, lane);
        v16bf a1 = ldfragA(lKpT + (mrow + 16) * KS + ks * 32, KS, lane);
        v16bf b0 = ldfragB(lVt + ks * 32, VS, lane);
        v16bf b1 = ldfragB(lVt + 16 * VS + ks * 32, VS, lane);
        acc[0] = wmma_bf(a0, b0, acc[0]);
        acc[1] = wmma_bf(a0, b1, acc[1]);
        acc[2] = wmma_bf(a1, b0, acc[2]);
        acc[3] = wmma_bf(a1, b1, acc[3]);
      }
    }
  }
  ksumb[((size_t)blockIdx.x * 64 + bh) * 256 + t] = ksacc;
  float* kvh = kvb + ((size_t)blockIdx.x * 64 + bh) * 8192;
  const int hf = lane >> 4;
#pragma unroll
  for (int mt = 0; mt < 2; mt++)
#pragma unroll
    for (int dt = 0; dt < 2; dt++) {
      v8f a = acc[mt * 2 + dt];
      int d = dt * 16 + (lane & 15);
#pragma unroll
      for (int v = 0; v < 8; v++) {
        int m = w * 32 + mt * 16 + hf * 8 + v;
        kvh[m * 32 + d] = a[v];
      }
    }
}

// ---------- qp = phi(q); out = (qp @ kv) * z ----------
__global__ __launch_bounds__(256)
void k_qattn(const bf16_t* __restrict__ qkv, const bf16_t* __restrict__ pw,
             const float* __restrict__ kvb, const float* __restrict__ ksumb,
             bf16_t* __restrict__ aout) {
  constexpr int PS = 32, US = 264, KV = 264;
  __shared__ __attribute__((aligned(16))) bf16_t lPK[8448];   // proj, later kv^T
  __shared__ __attribute__((aligned(16))) bf16_t lA[64 * PS];
  __shared__ __attribute__((aligned(16))) bf16_t lUQ[64 * US];  // u, later qp
  __shared__ float lKsum[256];
  __shared__ float lDiag[64];
  __shared__ float lPar[256];
  __shared__ float lRow[64];
  __shared__ float lZ[64];
  const int t = threadIdx.x, lane = t & 31, w = t >> 5;
  const int bh = blockIdx.y, b = bh >> 3, h = bh & 7;
  const int n0 = blockIdx.x * 64;
#pragma unroll
  for (int i = 0; i < 4; i++)
    *(v8bf*)(lPK + t * PS + i * 8) = *(const v8bf*)(pw + t * PS + i * 8);
  {
    int r = t >> 2, kg = (t & 3) * 8;
    v8bf qv = *(const v8bf*)(qkv + (size_t)(b * Nc + n0 + r) * 768 + h * 32 + kg);
    v8bf o;
#pragma unroll
    for (int i = 0; i < 8; i++) o[i] = f2bf(bf2f(qv[i]) * NRM);
    *(v8bf*)(lA + r * PS + kg) = o;
  }
  {
    float s = 0.f;
#pragma unroll
    for (int c = 0; c < 4; c++) s += ksumb[((size_t)c * 64 + bh) * 256 + t];
    lKsum[t] = s;
  }
  __syncthreads();
  if (t < 64) {
    float s = 0.f;
    for (int d = 0; d < 32; d++) { float xx = bf2f(lA[t * PS + d]); s += xx * xx; }
    lDiag[t] = 0.5f * s;
  }
  {
    const int rt = (w >> 1) * 16, cb = (w & 1) * 128;
    v16bf af = ldfragA(lA + rt * PS, PS, lane);
    const int hf = lane >> 4;
#pragma unroll
    for (int ct = 0; ct < 8; ct++) {
      const int mbase = cb + ct * 16;
      v16bf bf = ldfragB(lPK + mbase * PS, PS, lane);
      v8f u = wmma_bf(af, bf, z8());
      const int mcol = mbase + (lane & 15);
#pragma unroll
      for (int v = 0; v < 8; v++) lUQ[(rt + hf * 8 + v) * US + mcol] = f2bf(u[v]);
    }
  }
  __syncthreads();
  // stage kv^T (summing the 4 deterministic partial slices); proj no longer needed
  for (int d = 0; d < 32; d++) {
    float s = 0.f;
#pragma unroll
    for (int c = 0; c < 4; c++) s += kvb[(((size_t)c * 64 + bh) * 256 + t) * 32 + d];
    lPK[d * KV + t] = f2bf(s);
  }
  const int rr = t >> 2, seg = t & 3;
  {
    float mxp = -3.0e38f;
    for (int j = 0; j < 64; j++) mxp = fmaxf(mxp, bf2f(lUQ[rr * US + seg * 64 + j]));
    lPar[rr * 4 + seg] = mxp;
  }
  __syncthreads();
  if (t < 64)
    lRow[t] = fmaxf(fmaxf(lPar[t * 4], lPar[t * 4 + 1]), fmaxf(lPar[t * 4 + 2], lPar[t * 4 + 3]));
  __syncthreads();
  {
    const float dg = lDiag[rr], rm = lRow[rr];
    float zp = 0.f;
    for (int j = 0; j < 64; j++) {
      int m = seg * 64 + j;
      float u = bf2f(lUQ[rr * US + m]);
      float qp = (expf(u - dg - rm) + 1e-4f) * 0.0625f;
      lUQ[rr * US + m] = f2bf(qp);   // in-place: same thread owns read+write
      zp += qp * lKsum[m];
    }
    lPar[rr * 4 + seg] = zp;
  }
  __syncthreads();
  if (t < 64)
    lZ[t] = 1.0f / (lPar[t * 4] + lPar[t * 4 + 1] + lPar[t * 4 + 2] + lPar[t * 4 + 3] + 1e-6f);
  __syncthreads();
  {
    const int rt2 = (w >> 1) * 16, dt = (w & 1) * 16;
    v8f acc = z8();
#pragma unroll
    for (int ks = 0; ks < 8; ks++) {
      v16bf a = ldfragA(lUQ + rt2 * US + ks * 32, US, lane);
      v16bf bb = ldfragB(lPK + dt * KV + ks * 32, KV, lane);
      acc = wmma_bf(a, bb, acc);
    }
    const int hf = lane >> 4;
    const int d = dt + (lane & 15);
#pragma unroll
    for (int v = 0; v < 8; v++) {
      int rl = rt2 + hf * 8 + v;
      float val = acc[v] * lZ[rl];
      aout[(size_t)(b * Nc + n0 + rl) * 256 + h * 32 + d] = f2bf(val);
    }
  }
}

// ---------- host ----------
extern "C" void kernel_launch(void* const* d_in, const int* in_sizes, int n_in,
                              void* d_out, int out_size, void* d_ws, size_t ws_size,
                              hipStream_t stream) {
  (void)in_sizes; (void)n_in; (void)out_size; (void)ws_size;
  const float* x     = (const float*)d_in[0];
  const float* pos   = (const float*)d_in[1];
  const float* g1n   = (const float*)d_in[2];
  const float* b1n   = (const float*)d_in[3];
  const float* g2n   = (const float*)d_in[4];
  const float* b2n   = (const float*)d_in[5];
  const float* Wqkv  = (const float*)d_in[6];
  const float* Wproj = (const float*)d_in[7];
  const float* bproj = (const float*)d_in[8];
  const float* W1    = (const float*)d_in[9];
  const float* b1m   = (const float*)d_in[10];
  const float* W2    = (const float*)d_in[11];
  const float* b2m   = (const float*)d_in[12];
  const float* proj  = (const float*)d_in[13];

  char* ws = (char*)d_ws;
  size_t off = 0;
  auto take = [&](size_t bytes) -> void* {
    void* p = ws + off;
    off += (bytes + 255) & ~(size_t)255;
    return p;
  };
  float*    hbuf  = (float*)   take((size_t)ROWS * Dc * 4);
  bf16_t*   lnbuf = (bf16_t*)  take((size_t)ROWS * Dc * 2);
  bf16_t*   aobuf = (bf16_t*)  take((size_t)ROWS * Dc * 2);
  bf16_t*   big   = (bf16_t*)  take((size_t)ROWS * HIDc * 2);   // qkv (768) or mlp-hidden (1024)
  float*    kvb   = (float*)   take((size_t)4 * 64 * Mc * DHc * 4);
  float*    ksumb = (float*)   take((size_t)4 * 64 * Mc * 4);
  unsigned* umax  = (unsigned*)take(64 * 4);
  bf16_t*   wq_b  = (bf16_t*)  take((size_t)4 * Dc * 3 * Dc * 2);
  bf16_t*   wp_b  = (bf16_t*)  take((size_t)4 * Dc * Dc * 2);
  bf16_t*   w1_b  = (bf16_t*)  take((size_t)4 * Dc * HIDc * 2);
  bf16_t*   w2_b  = (bf16_t*)  take((size_t)4 * HIDc * Dc * 2);
  bf16_t*   pj_b  = (bf16_t*)  take((size_t)4 * Mc * DHc * 2);

  auto cvt = [&](const float* s, bf16_t* d, int n) {
    k_cvt<<<(n + 255) / 256, 256, 0, stream>>>(s, d, n);
  };
  cvt(Wqkv, wq_b, 4 * Dc * 3 * Dc);
  cvt(Wproj, wp_b, 4 * Dc * Dc);
  cvt(W1, w1_b, 4 * Dc * HIDc);
  cvt(W2, w2_b, 4 * HIDc * Dc);
  cvt(proj, pj_b, 4 * Mc * DHc);

  k_addpos<<<(ROWS * Dc) / 256, 256, 0, stream>>>(x, pos, hbuf);

  for (int l = 0; l < 4; ++l) {
    const bf16_t* wq = wq_b + (size_t)l * Dc * 3 * Dc;
    const bf16_t* wp = wp_b + (size_t)l * Dc * Dc;
    const bf16_t* w1 = w1_b + (size_t)l * Dc * HIDc;
    const bf16_t* w2 = w2_b + (size_t)l * HIDc * Dc;
    const bf16_t* pj = pj_b + (size_t)l * Mc * DHc;

    k_ln<<<ROWS, 256, 0, stream>>>(hbuf, g1n, b1n, lnbuf);
    k_gemm<0, 0><<<dim3(ROWS / 128, (3 * Dc) / 64), 256, 0, stream>>>(
        lnbuf, wq, nullptr, nullptr, big, Dc, 3 * Dc);
    k_zero_umax<<<1, 64, 0, stream>>>(umax);
    k_umax<<<dim3(Nc / 64, Bc * Hc), 256, 0, stream>>>(big, pj, umax);
    k_kvsum<<<dim3(4, Bc * Hc), 256, 0, stream>>>(big, pj, umax, kvb, ksumb);
    k_qattn<<<dim3(Nc / 64, Bc * Hc), 256, 0, stream>>>(big, pj, kvb, ksumb, aobuf);
    k_gemm<0, 1><<<dim3(ROWS / 128, Dc / 64), 256, 0, stream>>>(
        aobuf, wp, bproj + (size_t)l * Dc, hbuf, nullptr, Dc, Dc);
    k_ln<<<ROWS, 256, 0, stream>>>(hbuf, g2n, b2n, lnbuf);
    k_gemm<1, 0><<<dim3(ROWS / 128, HIDc / 64), 256, 0, stream>>>(
        lnbuf, w1, b1m + (size_t)l * HIDc, nullptr, big, Dc, HIDc);
    k_gemm<0, 1><<<dim3(ROWS / 128, Dc / 64), 256, 0, stream>>>(
        big, w2, b2m + (size_t)l * Dc, hbuf, nullptr, HIDc, Dc);
  }
  k_copy<<<(ROWS * Dc) / 256, 256, 0, stream>>>(hbuf, (float*)d_out);
}
